// QuadraticGNN_33492154974254
// MI455X (gfx1250) — matrix operand order repaired and checked by reading, compiled, not run on hardware
//
#include <hip/hip_runtime.h>

// ---------------------------------------------------------------------------
// QuadraticGNN on MI455X (gfx1250): 3x ResGatedGraphConv + mean-pool + MLP.
// Dense GEMMs run on v_wmma_f32_16x16x32_bf16 (f32 accumulate) with the A
// fragments register-resident across all weight matrices / N-tiles so HBM
// reads the activations exactly once per layer. K, co, and the number of
// fused weight matrices are compile-time template parameters so stores use
// immediate offsets and the N-tile loop can software-pipeline.
// ---------------------------------------------------------------------------

typedef __attribute__((ext_vector_type(16))) __bf16 v16bf;
typedef __attribute__((ext_vector_type(8)))  __bf16 bf16x8;
typedef __attribute__((ext_vector_type(8)))  float  v8f;

#define N_NODES  100000
#define N_EDGES  300000
#define N_GROUPS 32

struct Bias4 { const float* p[4]; };
struct Out4  { float*       p[4]; };

static __device__ __forceinline__ __bf16 f2bf(float f) {
  // round-to-nearest-even f32 -> bf16
  unsigned u = __builtin_bit_cast(unsigned, f);
  unsigned r = u + 0x7FFFu + ((u >> 16) & 1u);
  unsigned short h = (unsigned short)(r >> 16);
  return __builtin_bit_cast(__bf16, h);
}

// ---------------- elementwise casts ----------------------------------------

__global__ void cast_f32_bf16(const float* __restrict__ in,
                              __bf16* __restrict__ out, size_t n) {
  size_t i = (size_t)blockIdx.x * blockDim.x + threadIdx.x;
  if (i < n) out[i] = f2bf(in[i]);
}

// W [K, co] f32 row-major  ->  Wt [co, K] bf16 row-major (i.e. W^T)
__global__ void cast_wt(const float* __restrict__ W,
                        __bf16* __restrict__ Wt, int K, int co) {
  int i = blockIdx.x * blockDim.x + threadIdx.x;
  if (i >= K * co) return;
  int n = i / K, k = i % K;
  Wt[i] = f2bf(W[(size_t)k * co + n]);
}

__global__ void leaky_cast(const float* __restrict__ in,
                           __bf16* __restrict__ out, size_t n) {
  size_t i = (size_t)blockIdx.x * blockDim.x + threadIdx.x;
  if (i >= n) return;
  float v = in[i];
  v = v > 0.0f ? v : 0.01f * v;   // LeakyReLU(0.01)
  out[i] = f2bf(v);
}

__global__ void zero_f32(float* __restrict__ p, int n) {
  int i = blockIdx.x * blockDim.x + threadIdx.x;
  if (i < n) p[i] = 0.0f;
}

// ---------------- WMMA multi-GEMM -------------------------------------------
// For one 16-row M-tile per wave, computes NW outputs:
//   C_w[M,CO] = A[M,K](bf16) @ W_w[K,CO] + bias_w   (W_w staged as Wt[CO,K] bf16)
// A fragments (K/32 x v16bf) are loaded once into VGPRs and reused across all
// NW weight matrices and all CO/16 N-tiles; only the small L2-resident Wt
// streams through the inner loop (2 loads per v_wmma).
// Fragment layouts used here (wave32; contiguous K for A and B, not the ISA table order):
//   A 16x32 bf16 : lane m=lane&15, 16 contiguous K {base..base+15},
//                  base = (lane<16) ? 0 : 16 -> two 16B contiguous loads
//   B 32x16 bf16 : lane n=lane&15, 16 contiguous K of column n,
//                  K start = (lane<16) ? 0 : 16 -> one 32B load from Wt
//   C/D f32      : lane n=lane&15, VGPR r -> row r + ((lane<16)?0:8)
template <int K, int CO, int NW>
__global__ void gemm_multi_wmma(const __bf16* __restrict__ A,
                                const __bf16* __restrict__ WtAll, // [NW, CO, K]
                                Bias4 biases,                      // NW bias ptrs
                                Out4  Cs,                          // NW f32 out ptrs
                                __bf16* __restrict__ Cbf,          // optional (NW==1)
                                int M) {
  constexpr int KF = K / 32;
  constexpr int NT = CO / 16;
  const int lane = threadIdx.x & 31;
  const int wave = threadIdx.x >> 5;
  const int mt   = blockIdx.x * 8 + wave;
  if (mt * 16 >= M) return;                 // wave-uniform, WMMA-safe

  const int li    = lane & 15;
  const int hi16  = (lane >> 4) & 1;
  const int m     = mt * 16 + li;
  const int abase = hi16 * 16;
  const int bbase = hi16 * 16;

  // Load all A fragments for this M-tile into registers (<= 64 VGPRs).
  v16bf a[KF];
  const __bf16* arow = A + (size_t)m * K;
  #pragma unroll
  for (int f = 0; f < KF; ++f) {
    bf16x8 lo = *(const bf16x8*)(arow + f * 32 + abase);
    bf16x8 hi = *(const bf16x8*)(arow + f * 32 + 8 + abase);
    #pragma unroll
    for (int i = 0; i < 8; ++i) { a[f][i] = lo[i]; a[f][8 + i] = hi[i]; }
  }

  const int mrow0 = mt * 16 + hi16 * 8;
  #pragma unroll
  for (int w = 0; w < NW; ++w) {
    const __bf16* Wt   = WtAll + (size_t)w * CO * K;
    const float*  bias = biases.p[w];
    float*        Crow = Cs.p[w] + (size_t)mrow0 * CO + li;
    __bf16*       Brow = (NW == 1 && Cbf) ? (Cbf + (size_t)mrow0 * CO + li) : nullptr;
    #pragma unroll 2
    for (int nt = 0; nt < NT; ++nt) {
      const int n = nt * 16 + li;
      const __bf16* brow = Wt + (size_t)n * K + bbase;
      v8f c;
      #pragma unroll
      for (int r = 0; r < 8; ++r) c[r] = 0.0f;
      #pragma unroll
      for (int f = 0; f < KF; ++f) {
        v16bf b = *(const v16bf*)(brow + f * 32);
        c = __builtin_amdgcn_wmma_f32_16x16x32_bf16(
                /*neg_a=*/false, a[f], /*neg_b=*/false, b,
                /*c_mod=*/(short)0, c, /*reuse_a=*/false, /*reuse_b=*/false);
      }
      const float bv = bias[n];
      #pragma unroll
      for (int r = 0; r < 8; ++r) {
        float v = c[r] + bv;
        Crow[r * CO + nt * 16] = v;             // immediate-offset stores
        if (NW == 1 && Cbf) Brow[r * CO + nt * 16] = f2bf(v);
      }
    }
  }
}

// ---------------- edge phase ------------------------------------------------
// agg[dst] += relu(Xk[dst] + Xq[src]) * Xv[src]   (agg pre-seeded with h@Ws+bs)
__global__ void edge_kernel(const int* __restrict__ src,
                            const int* __restrict__ dst,
                            const float* __restrict__ Xk,
                            const float* __restrict__ Xq,
                            const float* __restrict__ Xv,
                            float* __restrict__ agg,
                            int E, int co) {
  size_t idx = (size_t)blockIdx.x * blockDim.x + threadIdx.x;
  const int chunks = co >> 2;
  size_t total = (size_t)E * chunks;
  if (idx >= total) return;
  int e  = (int)(idx / chunks);
  int c4 = (int)(idx % chunks) << 2;
  int s = src[e], d = dst[e];
  const float4 xk = *(const float4*)(Xk + (size_t)d * co + c4);
  const float4 xq = *(const float4*)(Xq + (size_t)s * co + c4);
  const float4 xv = *(const float4*)(Xv + (size_t)s * co + c4);
  float g0 = fmaxf(xk.x + xq.x, 0.0f) * xv.x;
  float g1 = fmaxf(xk.y + xq.y, 0.0f) * xv.y;
  float g2 = fmaxf(xk.z + xq.z, 0.0f) * xv.z;
  float g3 = fmaxf(xk.w + xq.w, 0.0f) * xv.w;
  float* a = agg + (size_t)d * co + c4;
  unsafeAtomicAdd(a + 0, g0);
  unsafeAtomicAdd(a + 1, g1);
  unsafeAtomicAdd(a + 2, g2);
  unsafeAtomicAdd(a + 3, g3);
}

// ---------------- pooling ---------------------------------------------------
__global__ void pool_kernel(const float* __restrict__ H,
                            const int* __restrict__ batch,
                            float* __restrict__ sums,   // [32,256]
                            float* __restrict__ cnt) {  // [32]
  size_t idx = (size_t)blockIdx.x * blockDim.x + threadIdx.x;
  size_t total = (size_t)N_NODES * 64;   // 256/4 chunks per node
  if (idx >= total) return;
  int node = (int)(idx >> 6);
  int c4   = (int)(idx & 63) << 2;
  int g = batch[node];
  const float4 h = *(const float4*)(H + (size_t)node * 256 + c4);
  float* s = sums + (size_t)g * 256 + c4;
  unsafeAtomicAdd(s + 0, h.x);
  unsafeAtomicAdd(s + 1, h.y);
  unsafeAtomicAdd(s + 2, h.z);
  unsafeAtomicAdd(s + 3, h.w);
  if (c4 == 0) unsafeAtomicAdd(cnt + g, 1.0f);
}

// ---------------- tiny pooled MLP (single block) ----------------------------
__global__ void mlp_kernel(const float* __restrict__ sums,
                           const float* __restrict__ cnt,
                           const float* __restrict__ W1, const float* __restrict__ b1,
                           const float* __restrict__ Wh, const float* __restrict__ bh,
                           const float* __restrict__ Wo, const float* __restrict__ bo,
                           const float* __restrict__ gamma, const float* __restrict__ beta,
                           const float* __restrict__ mean,  const float* __restrict__ var,
                           float* __restrict__ out) {
  __shared__ float P[32][256];
  __shared__ float H[32][64];
  __shared__ float H2[32][64];
  const int t = threadIdx.x;

  for (int i = t; i < 32 * 256; i += 256) {
    int g = i >> 8;
    P[g][i & 255] = sums[i] / fmaxf(cnt[g], 1.0f);
  }
  __syncthreads();

  // layer 0: 256 -> 64, bn[0], relu
  for (int i = t; i < 32 * 64; i += 256) {
    int g = i >> 6, j = i & 63;
    float acc = b1[j];
    for (int k = 0; k < 256; ++k) acc = fmaf(P[g][k], W1[k * 64 + j], acc);
    float inv = rsqrtf(var[j] + 1e-5f);
    float v = (acc - mean[j]) * (gamma[j] * inv) + beta[j];
    H[g][j] = fmaxf(v, 0.0f);
  }
  __syncthreads();

  // 3 hidden 64 -> 64 layers, bn[1..3], relu
  for (int l = 0; l < 3; ++l) {
    const float* Wl = Wh + l * 64 * 64;
    const float* bl = bh + l * 64;
    const float* ga = gamma + (l + 1) * 64;
    const float* be = beta  + (l + 1) * 64;
    const float* me = mean  + (l + 1) * 64;
    const float* va = var   + (l + 1) * 64;
    for (int i = t; i < 32 * 64; i += 256) {
      int g = i >> 6, j = i & 63;
      float acc = bl[j];
      for (int k = 0; k < 64; ++k) acc = fmaf(H[g][k], Wl[k * 64 + j], acc);
      float inv = rsqrtf(va[j] + 1e-5f);
      float v = (acc - me[j]) * (ga[j] * inv) + be[j];
      H2[g][j] = fmaxf(v, 0.0f);
    }
    __syncthreads();
    for (int i = t; i < 32 * 64; i += 256) { int g = i >> 6, j = i & 63; H[g][j] = H2[g][j]; }
    __syncthreads();
  }

  // output: 64 -> 8
  for (int i = t; i < 32 * 8; i += 256) {
    int g = i >> 3, j = i & 7;
    float acc = bo[j];
    for (int k = 0; k < 64; ++k) acc = fmaf(H[g][k], Wo[k * 8 + j], acc);
    out[i] = acc;
  }
}

// ---------------- orchestration ---------------------------------------------

static inline unsigned cdiv_sz(size_t a, unsigned b) {
  return (unsigned)((a + b - 1) / b);
}

extern "C" void kernel_launch(void* const* d_in, const int* in_sizes, int n_in,
                              void* d_out, int out_size, void* d_ws, size_t ws_size,
                              hipStream_t stream) {
  (void)in_sizes; (void)n_in; (void)out_size; (void)ws_size;

  const float* x     = (const float*)d_in[0];
  const int*   ei    = (const int*)d_in[1];     // [2, E]: src row then dst row
  const int*   batch = (const int*)d_in[2];
  const int*   src   = ei;
  const int*   dst   = ei + N_EDGES;

  const float* mlp_W1 = (const float*)d_in[33];
  const float* mlp_b1 = (const float*)d_in[34];
  const float* mlp_Wh = (const float*)d_in[35];
  const float* mlp_bh = (const float*)d_in[36];
  const float* mlp_Wo = (const float*)d_in[37];
  const float* mlp_bo = (const float*)d_in[38];
  const float* bn_g   = (const float*)d_in[39];
  const float* bn_b   = (const float*)d_in[40];
  const float* bn_m   = (const float*)d_in[41];
  const float* bn_v   = (const float*)d_in[42];

  // workspace carve-up
  char* p = (char*)d_ws;
  auto take = [&](size_t bytes) -> char* {
    char* r = p;
    p += (bytes + 255) & ~(size_t)255;
    return r;
  };
  __bf16* ABF  = (__bf16*)take((size_t)N_NODES * 256 * 2); // current layer bf16 input
  __bf16* H2BF = (__bf16*)take((size_t)N_NODES * 256 * 2); // leaky(h1) in bf16
  float*  XK   = (float*) take((size_t)N_NODES * 256 * 4);
  float*  XQ   = (float*) take((size_t)N_NODES * 256 * 4);
  float*  XV   = (float*) take((size_t)N_NODES * 256 * 4);
  float*  AGG  = (float*) take((size_t)N_NODES * 256 * 4); // h@Ws+bs, then += messages
  float*  HOUT = (float*) take((size_t)N_NODES * 256 * 4); // layer output (fp32)
  __bf16* WT4  = (__bf16*)take((size_t)4 * 256 * 256 * 2); // transposed weight staging
  float*  POOL = (float*) take((size_t)(32 * 256 + 32) * 4);
  float*  CNT  = POOL + 32 * 256;

  // bf16 copy of x
  {
    size_t n = (size_t)N_NODES * 128;
    cast_f32_bf16<<<cdiv_sz(n, 256), 256, 0, stream>>>(x, ABF, n);
  }

  const int dims[4] = {128, 64, 128, 256};
  const int mtiles = N_NODES / 16;                 // 6250 exact
  const dim3 ggrid((unsigned)((mtiles + 7) / 8));

  // fused 4-matrix GEMM (Wk, Wq, Wv, Ws share the same A)
  auto launch4 = [&](int K, int co, const float* const W[4],
                     const float* const b[4], float* const C[4]) {
    for (int w = 0; w < 4; ++w)
      cast_wt<<<cdiv_sz((size_t)(K * co), 256), 256, 0, stream>>>(
          W[w], WT4 + (size_t)w * K * co, K, co);
    Bias4 bb = {{b[0], b[1], b[2], b[3]}};
    Out4  cc = {{C[0], C[1], C[2], C[3]}};
    if (K == 128 && co == 64) {
      gemm_multi_wmma<128, 64, 4><<<ggrid, 256, 0, stream>>>(ABF, WT4, bb, cc,
                                                             nullptr, N_NODES);
    } else if (K == 64 && co == 128) {
      gemm_multi_wmma<64, 128, 4><<<ggrid, 256, 0, stream>>>(ABF, WT4, bb, cc,
                                                             nullptr, N_NODES);
    } else {  // K == 128, co == 256
      gemm_multi_wmma<128, 256, 4><<<ggrid, 256, 0, stream>>>(ABF, WT4, bb, cc,
                                                              nullptr, N_NODES);
    }
  };

  // single GEMM with f32 + bf16 outputs (Wl projection, K == co)
  auto launch1 = [&](int K, const __bf16* A, const float* W,
                     const float* b, float* C, __bf16* Cbf) {
    cast_wt<<<cdiv_sz((size_t)(K * K), 256), 256, 0, stream>>>(W, WT4, K, K);
    Bias4 bb = {{b, b, b, b}};
    Out4  cc = {{C, C, C, C}};
    if (K == 64) {
      gemm_multi_wmma<64, 64, 1><<<ggrid, 256, 0, stream>>>(A, WT4, bb, cc, Cbf,
                                                            N_NODES);
    } else if (K == 128) {
      gemm_multi_wmma<128, 128, 1><<<ggrid, 256, 0, stream>>>(A, WT4, bb, cc, Cbf,
                                                              N_NODES);
    } else {  // K == 256
      gemm_multi_wmma<256, 256, 1><<<ggrid, 256, 0, stream>>>(A, WT4, bb, cc, Cbf,
                                                              N_NODES);
    }
  };

  for (int l = 0; l < 3; ++l) {
    const int ci = dims[l], co = dims[l + 1];
    const int base = 3 + l * 10;
    const float* W4[4] = {(const float*)d_in[base + 0], (const float*)d_in[base + 1],
                          (const float*)d_in[base + 2], (const float*)d_in[base + 3]};
    const float* b4[4] = {(const float*)d_in[base + 4], (const float*)d_in[base + 5],
                          (const float*)d_in[base + 6], (const float*)d_in[base + 7]};
    const float* Wl = (const float*)d_in[base + 8];
    const float* bl = (const float*)d_in[base + 9];
    float* C4[4] = {XK, XQ, XV, AGG};   // AGG seeded with h@Ws+bs

    launch4(ci, co, W4, b4, C4);

    {
      size_t total = (size_t)N_EDGES * (co / 4);
      edge_kernel<<<cdiv_sz(total, 256), 256, 0, stream>>>(src, dst, XK, XQ, XV,
                                                           AGG, N_EDGES, co);
    }
    {
      size_t n = (size_t)N_NODES * co;
      leaky_cast<<<cdiv_sz(n, 256), 256, 0, stream>>>(AGG, H2BF, n);
    }
    // h_out = leaky(h1) @ Wl + bl ; fp32 -> HOUT, bf16 -> ABF (next layer input)
    launch1(co, H2BF, Wl, bl, HOUT, ABF);
  }

  // global mean pool over batch ids (layer-3 co == 256)
  zero_f32<<<cdiv_sz((size_t)(32 * 256 + 32), 256), 256, 0, stream>>>(POOL, 32 * 256 + 32);
  {
    size_t total = (size_t)N_NODES * 64;
    pool_kernel<<<cdiv_sz(total, 256), 256, 0, stream>>>(HOUT, batch, POOL, CNT);
  }

  mlp_kernel<<<1, 256, 0, stream>>>(POOL, CNT, mlp_W1, mlp_b1, mlp_Wh, mlp_bh,
                                    mlp_Wo, mlp_bo, bn_g, bn_b, bn_m, bn_v,
                                    (float*)d_out);
}
